// Mamba3Predictor_88536455840130
// MI455X (gfx1250) — matrix operand order repaired and checked by reading, not compiled, not run on hardware
//
#include <hip/hip_runtime.h>

typedef __attribute__((ext_vector_type(16))) __bf16 v16bf;
typedef __attribute__((ext_vector_type(8)))  float  v8f;

// ---- problem constants (from setup_inputs) ----
static constexpr int Bb   = 4;
static constexpr int NV   = 256;
static constexpr int NQ   = 32;
static constexpr int Tt   = 288;           // NV + NQ
static constexpr int Dd   = 1024;
static constexpr int DI   = 2048;
static constexpr int Hh   = 32;
static constexpr int Nn   = 128;
static constexpr int Pp   = 64;            // DI / H
static constexpr int Ee   = 1536;
static constexpr int NL   = 12;
static constexpr int PROJ = 2*DI + 2*Nn + Hh;  // 4384
static constexpr int ROWS = Bb * Tt;           // 1152

__device__ __forceinline__ unsigned short f2bf(float f) {
    unsigned int u = __float_as_uint(f);
    u += 0x7FFFu + ((u >> 16) & 1u);           // round to nearest even
    return (unsigned short)(u >> 16);
}

// =====================================================================
// WMMA GEMM: C[MxN] = A(bf16, MxK) * W(f32->bf16, KxN) (+bias) (+addend)
// 128x128 block tile, 256 threads / 8 waves (wave32); each wave computes
// a 32x64 patch as 2x4 v_wmma_f32_16x16x32_bf16 per 32-wide K step.
// Double-buffered LDS; A tile staged via GLOBAL_LOAD_ASYNC_TO_LDS_B128
// (ASYNCcnt); W tile converted f32->bf16 with one v_perm_b32 per K-pair
// and stored transposed so B fragments gather contiguous K-pairs.
// =====================================================================
#define TM 128
#define TN 128
#define TK 32
#define LSTR 40   // LDS row stride in halves (80B -> rows stay 16B-aligned)

__global__ __launch_bounds__(256)
void gemm_bf16f32_kernel(const unsigned short* __restrict__ A,
                         const float* __restrict__ W,
                         const float* __restrict__ bias,
                         const float* addend,
                         float* C,
                         int M, int N, int K)
{
    __shared__ unsigned short As[2][TM * LSTR];
    __shared__ unsigned short Bs[2][TN * LSTR];   // transposed: [n][k]

    const int tid  = threadIdx.x;
    const int lane = tid & 31;
    const int wv   = tid >> 5;        // wave 0..7
    const int hb   = lane >> 4;       // half-wave
    const int idx  = lane & 15;

    const int m0 = blockIdx.y * TM;
    const int n0 = blockIdx.x * TN;
    const int wm = (wv >> 1) * 32;    // 0,32,64,96
    const int wn = (wv & 1) * 64;     // 0,64

    v8f acc[2][4];
    #pragma unroll
    for (int i = 0; i < 2; ++i)
        #pragma unroll
        for (int j = 0; j < 4; ++j)
            #pragma unroll
            for (int r = 0; r < 8; ++r) acc[i][j][r] = 0.f;

    const int aRow = tid >> 1;          // 0..127
    const int aSeg = (tid & 1) * 16;    // 0 or 16 halves
    const int wkp  = tid & 15;          // K-pair id: k = 2*wkp, 2*wkp+1
    const int wng  = tid >> 4;          // 0..15: 8-column group

    const bool aValid = (m0 + aRow) < M;
    const bool wFast  = (n0 + TN) <= N;

    // ---- async A tile stage: 16B per lane straight into LDS ----
    auto issueA = [&](int buf, int k0) {
        if (aValid) {
            unsigned ldsa = (unsigned)(unsigned long long)
                            &As[buf][aRow * LSTR + aSeg];
            unsigned long long ga = (unsigned long long)
                            (A + (size_t)(m0 + aRow) * K + k0 + aSeg);
            asm volatile("global_load_async_to_lds_b128 %0, %1, off"
                         :: "v"(ldsa), "v"(ga) : "memory");
        }
    };

    // ---- W tile: two K-adjacent rows x 8 cols into registers ----
    auto loadW = [&](int k0, float* r0, float* r1) {
        const int k = k0 + 2 * wkp;
        if (wFast) {
            const float4* p0 = reinterpret_cast<const float4*>(
                W + (size_t)k * N + n0 + wng * 8);
            const float4* p1 = reinterpret_cast<const float4*>(
                W + (size_t)(k + 1) * N + n0 + wng * 8);
            float4 a0 = p0[0], a1 = p0[1];
            float4 b0 = p1[0], b1 = p1[1];
            r0[0] = a0.x; r0[1] = a0.y; r0[2] = a0.z; r0[3] = a0.w;
            r0[4] = a1.x; r0[5] = a1.y; r0[6] = a1.z; r0[7] = a1.w;
            r1[0] = b0.x; r1[1] = b0.y; r1[2] = b0.z; r1[3] = b0.w;
            r1[4] = b1.x; r1[5] = b1.y; r1[6] = b1.z; r1[7] = b1.w;
        } else {
            #pragma unroll
            for (int c = 0; c < 8; ++c) {
                int n = n0 + wng * 8 + c;
                r0[c] = (n < N) ? W[(size_t)k * N + n]       : 0.f;
                r1[c] = (n < N) ? W[(size_t)(k + 1) * N + n] : 0.f;
            }
        }
    };

    // ---- pack two bf16 (K-pair) with v_perm, one b32 store per column ----
    auto storeW = [&](int buf, const float* r0, const float* r1) {
        #pragma unroll
        for (int c = 0; c < 8; ++c) {
            unsigned lo = __float_as_uint(r0[c]) + 0x8000u;   // round-half-up
            unsigned hi = __float_as_uint(r1[c]) + 0x8000u;
            unsigned u  = __builtin_amdgcn_perm(hi, lo, 0x07060302u);
            *reinterpret_cast<unsigned*>(
                &Bs[buf][(wng * 8 + c) * LSTR + 2 * wkp]) = u;
        }
    };

    // ---- fragment gather (ISA 7.12.2 layouts) + 8 WMMA ----
    auto compute = [&](int buf) {
        union Frag { unsigned int u[8]; v16bf v; } af[2], bfrag[4];
        #pragma unroll
        for (int i = 0; i < 2; ++i) {
            const unsigned short* ar = &As[buf][(wm + i * 16 + idx) * LSTR];
            #pragma unroll
            for (int j = 0; j < 8; ++j) {
                int kp = ((j < 4) ? 2 * j : 2 * j + 8) + hb * 8;   // A map
                af[i].u[j] = *reinterpret_cast<const unsigned int*>(ar + kp);
            }
        }
        #pragma unroll
        for (int i = 0; i < 4; ++i) {
            const unsigned short* br = &Bs[buf][(wn + i * 16 + idx) * LSTR];
            #pragma unroll
            for (int j = 0; j < 8; ++j) {
                int kp = ((j < 4) ? 2 * j : 2 * j + 8) + hb * 8;   // B map
                bfrag[i].u[j] = *reinterpret_cast<const unsigned int*>(br + kp);
            }
        }
        #pragma unroll
        for (int i = 0; i < 2; ++i)
            #pragma unroll
            for (int j = 0; j < 4; ++j)
                acc[i][j] = __builtin_amdgcn_wmma_f32_16x16x32_bf16(
                    false, af[i].v, false, bfrag[j].v, (short)0, acc[i][j],
                    false, false);
    };

    // ---- software pipeline over K ----
    float w0[8], w1[8];
    issueA(0, 0);
    loadW(0, w0, w1);
    storeW(0, w0, w1);
    asm volatile("s_wait_asynccnt 0x0" ::: "memory");
    __syncthreads();

    const int nsteps = K / TK;
    for (int s = 1; s < nsteps; ++s) {
        const int buf = s & 1;
        issueA(buf, s * TK);          // prefetch next A (ASYNCcnt)
        loadW(s * TK, w0, w1);        // prefetch next W into VGPRs (LOADcnt)
        compute(buf ^ 1);             // WMMA on current tile overlaps copies
        storeW(buf, w0, w1);
        asm volatile("s_wait_asynccnt 0x0" ::: "memory");
        __syncthreads();
    }
    compute((nsteps - 1) & 1);

    // ---- epilogue: C/D layout M = r + half*8, N = idx ----
    #pragma unroll
    for (int i = 0; i < 2; ++i)
        #pragma unroll
        for (int j = 0; j < 4; ++j)
            #pragma unroll
            for (int r = 0; r < 8; ++r) {
                int row = m0 + wm + i * 16 + hb * 8 + r;
                int col = n0 + wn + j * 16 + idx;
                if (row < M && col < N) {
                    float v = acc[i][j][r];
                    if (bias)   v += bias[col];
                    if (addend) v += addend[(size_t)row * N + col];
                    C[(size_t)row * N + col] = v;
                }
            }
}

// =====================================================================
// elementwise / norm kernels
// =====================================================================
__global__ void cvt_bf16_kernel(const float* __restrict__ in,
                                unsigned short* __restrict__ out_, int n)
{
    int id = blockIdx.x * blockDim.x + threadIdx.x;
    if (id < n) out_[id] = f2bf(in[id]);
}

__global__ __launch_bounds__(256)
void rmsnorm_kernel(const float* __restrict__ in, const float* __restrict__ w,
                    unsigned short* out_bf, float* out_f, int D)
{
    int row = blockIdx.x;
    const float* x = in + (size_t)row * D;
    __shared__ float red[256];
    float ss = 0.f;
    for (int d = threadIdx.x; d < D; d += 256) { float v = x[d]; ss += v * v; }
    red[threadIdx.x] = ss;
    __syncthreads();
    for (int st = 128; st > 0; st >>= 1) {
        if (threadIdx.x < st) red[threadIdx.x] += red[threadIdx.x + st];
        __syncthreads();
    }
    float rinv = rsqrtf(red[0] / (float)D + 1e-5f);
    for (int d = threadIdx.x; d < D; d += 256) {
        float v = x[d] * rinv * w[d];
        if (out_bf) out_bf[(size_t)row * D + d] = f2bf(v);
        if (out_f)  out_f [(size_t)row * D + d] = v;
    }
}

__global__ void prep_dt_kernel(const float* __restrict__ proj,
                               const float* __restrict__ dt_bias,
                               float* __restrict__ dtbuf, int total)
{
    int id = blockIdx.x * blockDim.x + threadIdx.x;
    if (id >= total) return;
    int tok = id >> 5, h = id & 31;
    float v = proj[(size_t)tok * PROJ + 2 * DI + 2 * Nn + h] + dt_bias[h];
    dtbuf[id] = (v > 20.f) ? v : log1pf(expf(v));
}

// =====================================================================
// SSD scan: one block per (b,h); thread = (p, n-quarter); 32 state floats
// per thread kept in VGPRs; shifted (t-1) operands carried in registers.
// =====================================================================
__global__ __launch_bounds__(256)
void ssd_scan_kernel(const float* __restrict__ proj, const float* __restrict__ dtbuf,
                     const float* __restrict__ A_log, const float* __restrict__ D_h,
                     float* __restrict__ y)
{
    const int bh = blockIdx.x;
    const int b  = bh >> 5;
    const int h  = bh & 31;
    const int p  = threadIdx.x >> 2;
    const int nq = threadIdx.x & 3;
    const float Ah = -expf(A_log[h]);
    const float Dh = D_h[h];

    float s[32], Bprev[32];
    #pragma unroll
    for (int j = 0; j < 32; ++j) { s[j] = 0.f; Bprev[j] = 0.f; }
    float xprev = 0.f, dtprev = 0.f;

    for (int t = 0; t < Tt; ++t) {
        const int tok = b * Tt + t;
        const float* prow = proj + (size_t)tok * PROJ;
        float dtc = dtbuf[tok * Hh + h];
        float da  = expf(dtc * Ah);
        float xv  = prow[DI + h * Pp + p];

        float Bv[32], Cv[32];
        const float4* B4 = reinterpret_cast<const float4*>(prow + 2 * DI + nq * 32);
        const float4* C4 = reinterpret_cast<const float4*>(prow + 2 * DI + Nn + nq * 32);
        #pragma unroll
        for (int q = 0; q < 8; ++q) {
            float4 bb = B4[q]; float4 cc = C4[q];
            Bv[4*q+0] = bb.x; Bv[4*q+1] = bb.y; Bv[4*q+2] = bb.z; Bv[4*q+3] = bb.w;
            Cv[4*q+0] = cc.x; Cv[4*q+1] = cc.y; Cv[4*q+2] = cc.z; Cv[4*q+3] = cc.w;
        }

        float cu  = dtc    * xv;      // u  = dt*x*B
        float cup = dtprev * xprev;   // up = dt_prev*x_prev*B_prev
        float yacc = 0.f;
        #pragma unroll
        for (int j = 0; j < 32; ++j) {
            float sv = da * (s[j] + 0.5f * (cup * Bprev[j])) + 0.5f * (cu * Bv[j]);
            s[j] = sv;
            yacc += sv * Cv[j];
        }
        // reduce the n-dimension across the 4 lanes of this p (wave32 shuffles)
        yacc += __shfl_xor(yacc, 1, 32);
        yacc += __shfl_xor(yacc, 2, 32);
        if (nq == 0)
            y[(size_t)tok * DI + h * Pp + p] = yacc + Dh * xv;

        #pragma unroll
        for (int j = 0; j < 32; ++j) Bprev[j] = Bv[j];
        xprev = xv; dtprev = dtc;
    }
}

__global__ __launch_bounds__(256)
void gated_rmsnorm_kernel(const float* __restrict__ y, const float* __restrict__ proj,
                          const float* __restrict__ gw, unsigned short* __restrict__ outbf)
{
    int tok = blockIdx.x;
    const float* yr = y + (size_t)tok * DI;
    const float* zr = proj + (size_t)tok * PROJ;     // z slice = [0, DI)
    __shared__ float red[256];
    float g[8];
    float ss = 0.f;
    #pragma unroll
    for (int q = 0; q < 8; ++q) {
        int d = threadIdx.x + q * 256;
        float z = zr[d];
        float v = yr[d] * (z / (1.f + expf(-z)));    // y * silu(z)
        g[q] = v;
        ss += v * v;
    }
    red[threadIdx.x] = ss;
    __syncthreads();
    for (int st = 128; st > 0; st >>= 1) {
        if (threadIdx.x < st) red[threadIdx.x] += red[threadIdx.x + st];
        __syncthreads();
    }
    float rinv = rsqrtf(red[0] / (float)DI + 1e-5f);
    #pragma unroll
    for (int q = 0; q < 8; ++q) {
        int d = threadIdx.x + q * 256;
        outbf[(size_t)tok * DI + d] = f2bf(g[q] * rinv * gw[d]);
    }
}

__global__ void pool_kernel(const float* __restrict__ hn, float* __restrict__ pooled)
{
    int id = blockIdx.x * blockDim.x + threadIdx.x;
    if (id >= Bb * Dd) return;
    int b = id >> 10, d = id & 1023;
    float sum = 0.f;
    for (int t = 0; t < Tt; ++t)
        sum += hn[((size_t)b * Tt + t) * Dd + d];
    pooled[id] = sum * (1.f / (float)Tt);
}

__global__ void head_kernel(const float* __restrict__ pooled, const float* __restrict__ w,
                            const float* __restrict__ bias, float* __restrict__ out_)
{
    int id = blockIdx.x * blockDim.x + threadIdx.x;
    if (id >= Bb * Ee) return;
    int b = id / Ee, e = id % Ee;
    const float* pv = pooled + (size_t)b * Dd;
    float acc = bias[e];
    for (int d = 0; d < Dd; ++d)
        acc += pv[d] * w[(size_t)d * Ee + e];
    out_[id] = acc;
}

__global__ __launch_bounds__(256)
void l2norm_kernel(float* out_)
{
    int b = blockIdx.x;
    __shared__ float red[256];
    float ss = 0.f;
    for (int e = threadIdx.x; e < Ee; e += 256) {
        float v = out_[(size_t)b * Ee + e];
        ss += v * v;
    }
    red[threadIdx.x] = ss;
    __syncthreads();
    for (int st = 128; st > 0; st >>= 1) {
        if (threadIdx.x < st) red[threadIdx.x] += red[threadIdx.x + st];
        __syncthreads();
    }
    float inv = 1.f / fmaxf(sqrtf(red[0]), 1e-12f);
    for (int e = threadIdx.x; e < Ee; e += 256)
        out_[(size_t)b * Ee + e] *= inv;
}

// =====================================================================
// launch
// =====================================================================
extern "C" void kernel_launch(void* const* d_in, const int* in_sizes, int n_in,
                              void* d_out, int out_size, void* d_ws, size_t ws_size,
                              hipStream_t stream)
{
    (void)in_sizes; (void)n_in; (void)out_size; (void)ws_size;
    const float* visual   = (const float*)d_in[0];
    const float* query    = (const float*)d_in[1];
    const float* vis_w    = (const float*)d_in[2];
    const float* vis_b    = (const float*)d_in[3];
    const float* qry_w    = (const float*)d_in[4];
    const float* qry_b    = (const float*)d_in[5];
    const float* norm_ws  = (const float*)d_in[6];
    const float* in_ws    = (const float*)d_in[7];
    const float* dt_bias  = (const float*)d_in[8];
    const float* A_logs   = (const float*)d_in[9];
    const float* D_ssm    = (const float*)d_in[10];
    const float* gnorm_ws = (const float*)d_in[11];
    const float* out_ws   = (const float*)d_in[12];
    const float* norm_f_w = (const float*)d_in[13];
    const float* head_w   = (const float*)d_in[14];
    const float* head_b   = (const float*)d_in[15];
    float* out = (float*)d_out;

    char* ws = (char*)d_ws;
    auto carve = [&](size_t bytes) -> char* {
        char* p = ws;
        ws += (bytes + 255) & ~(size_t)255;
        return p;
    };
    unsigned short* visbf  = (unsigned short*)carve((size_t)Bb * NV * Dd * 2);
    unsigned short* qrybf  = (unsigned short*)carve((size_t)Bb * NQ * Dd * 2);
    float*          hidden = (float*)         carve((size_t)ROWS * Dd * 4);
    unsigned short* xin    = (unsigned short*)carve((size_t)ROWS * Dd * 2);
    float*          proj   = (float*)         carve((size_t)ROWS * PROJ * 4);
    float*          dtbuf  = (float*)         carve((size_t)ROWS * Hh * 4);
    float*          ybuf   = (float*)         carve((size_t)ROWS * DI * 4);
    unsigned short* gbf    = (unsigned short*)carve((size_t)ROWS * DI * 2);
    float*          hn     = (float*)         carve((size_t)ROWS * Dd * 4);
    float*          pooled = (float*)         carve((size_t)Bb * Dd * 4);

    // 1. inputs -> bf16
    {
        int n1 = Bb * NV * Dd;
        cvt_bf16_kernel<<<(n1 + 255) / 256, 256, 0, stream>>>(visual, visbf, n1);
        int n2 = Bb * NQ * Dd;
        cvt_bf16_kernel<<<(n2 + 255) / 256, 256, 0, stream>>>(query, qrybf, n2);
    }

    // 2. input projections -> hidden (B, 288, D)
    for (int b = 0; b < Bb; ++b) {
        dim3 g1((Dd + TN - 1) / TN, (NV + TM - 1) / TM);
        gemm_bf16f32_kernel<<<g1, 256, 0, stream>>>(
            visbf + (size_t)b * NV * Dd, vis_w, vis_b, nullptr,
            hidden + (size_t)b * Tt * Dd, NV, Dd, Dd);
        dim3 g2((Dd + TN - 1) / TN, (NQ + TM - 1) / TM);
        gemm_bf16f32_kernel<<<g2, 256, 0, stream>>>(
            qrybf + (size_t)b * NQ * Dd, qry_w, qry_b, nullptr,
            hidden + ((size_t)b * Tt + NV) * Dd, NQ, Dd, Dd);
    }

    // 3. layers
    for (int L = 0; L < NL; ++L) {
        rmsnorm_kernel<<<ROWS, 256, 0, stream>>>(
            hidden, norm_ws + (size_t)L * Dd, xin, nullptr, Dd);

        dim3 gp((PROJ + TN - 1) / TN, (ROWS + TM - 1) / TM);
        gemm_bf16f32_kernel<<<gp, 256, 0, stream>>>(
            xin, in_ws + (size_t)L * Dd * PROJ, nullptr, nullptr,
            proj, ROWS, PROJ, Dd);

        int td = ROWS * Hh;
        prep_dt_kernel<<<(td + 255) / 256, 256, 0, stream>>>(
            proj, dt_bias + (size_t)L * Hh, dtbuf, td);

        ssd_scan_kernel<<<Bb * Hh, 256, 0, stream>>>(
            proj, dtbuf, A_logs + (size_t)L * Hh, D_ssm + (size_t)L * Hh, ybuf);

        gated_rmsnorm_kernel<<<ROWS, 256, 0, stream>>>(
            ybuf, proj, gnorm_ws + (size_t)L * DI, gbf);

        dim3 go((Dd + TN - 1) / TN, (ROWS + TM - 1) / TM);
        gemm_bf16f32_kernel<<<go, 256, 0, stream>>>(
            gbf, out_ws + (size_t)L * DI * Dd, nullptr, hidden,
            hidden, ROWS, Dd, DI);
    }

    // 4. head
    rmsnorm_kernel<<<ROWS, 256, 0, stream>>>(hidden, norm_f_w, nullptr, hn, Dd);
    pool_kernel<<<(Bb * Dd + 255) / 256, 256, 0, stream>>>(hn, pooled);
    head_kernel<<<(Bb * Ee + 255) / 256, 256, 0, stream>>>(pooled, head_w, head_b, out);
    l2norm_kernel<<<Bb, 256, 0, stream>>>(out);
}
